// AnalyseAttention_21964462752055
// MI455X (gfx1250) — matrix operand order repaired and checked by reading, not compiled, not run on hardware
//
#include <hip/hip_runtime.h>
#include <hip/hip_bf16.h>

typedef __bf16 bf16_t;
typedef __attribute__((ext_vector_type(16))) __bf16 v16bf;
typedef __attribute__((ext_vector_type(8)))  float  v8f;

#define BD    8
#define NSEQ  1024
#define CDIM  768
#define NH    12
#define HD    64
#define MTOK  (BD * NSEQ)     /* 8192 */
#define QKVN  (3 * CDIM)      /* 2304 */

#ifndef __has_builtin
#define __has_builtin(x) 0
#endif
#if defined(__AMDGCN__) &&                                         \
    __has_builtin(__builtin_amdgcn_global_load_async_to_lds_b128) && \
    __has_builtin(__builtin_amdgcn_s_wait_asynccnt)
#define HAVE_ASYNC_LDS 1
#else
#define HAVE_ASYNC_LDS 0
#endif

union FragU { uint4 u[2]; v16bf v; };

#if HAVE_ASYNC_LDS
typedef __attribute__((ext_vector_type(4))) int v4i;
typedef __attribute__((address_space(1))) v4i gv4i_t;  /* global int4 */
typedef __attribute__((address_space(3))) v4i lv4i_t;  /* LDS int4 */
/* Generic LDS pointer = {aperture_hi, lds_offset} (ISA 10.2), so the low 32
 * bits are exactly the DS offset; global/flat share address values. */
__device__ __forceinline__ gv4i_t* as1(const void* p) {
  return (gv4i_t*)(unsigned long long)p;
}
__device__ __forceinline__ lv4i_t* as3(void* p) {
  return (lv4i_t*)(unsigned)(unsigned long long)p;
}
#endif

/* ---------------- fp32 -> bf16 conversion ---------------- */
__global__ void f32_to_bf16_kernel(const float* __restrict__ in,
                                   bf16_t* __restrict__ out, int n) {
  int i = blockIdx.x * blockDim.x + threadIdx.x;
  if (i < n) out[i] = (bf16_t)in[i];
}

/* ----------------------------------------------------------
 * Generic bf16 WMMA GEMM:  C[m][n] = sum_k A[m][k] * W[n][k]
 * Block: 256 threads = 8 waves. Tile BM=128, BN=64, BK=32.
 * Each wave: 16 rows x 64 cols = 4 f32 16x16 accumulators.
 * Tiles staged via async global->LDS copies when available.
 * mode 0: outF[m*CDIM+n] = acc + bias[n]        (proj)
 * mode 1: scatter to Q(B,H,N,D), K(B,H,N,D), V^T(B,H,D,N) bf16
 * ---------------------------------------------------------- */
__global__ void __launch_bounds__(256)
gemm_bf16_kernel(const bf16_t* __restrict__ A, const bf16_t* __restrict__ W,
                 int Kdim, int mode,
                 float* __restrict__ outF, const float* __restrict__ bias,
                 bf16_t* __restrict__ qb, bf16_t* __restrict__ kb,
                 bf16_t* __restrict__ vtb) {
  __shared__ bf16_t sA[128][40];   /* +8 pad vs 32 to dodge bank conflicts */
  __shared__ bf16_t sB[64][40];

  const int tid  = threadIdx.x;
  const int lane = tid & 31;
  const int wave = tid >> 5;
  const int m0   = blockIdx.y * 128;
  const int n0   = blockIdx.x * 64;

  v8f acc[4];
#pragma unroll
  for (int g = 0; g < 4; ++g)
#pragma unroll
    for (int r = 0; r < 8; ++r) acc[g][r] = 0.0f;

  for (int kt = 0; kt < Kdim; kt += 32) {
    __syncthreads();
#if HAVE_ASYNC_LDS
    { /* A tile 128x32: two async b128 transfers per thread */
      int idx = tid * 16;
      int r = idx >> 5, c = idx & 31;
      const bf16_t* g = A + (size_t)(m0 + r) * Kdim + kt + c;
      __builtin_amdgcn_global_load_async_to_lds_b128(as1(g),
                                                     as3(&sA[r][c]), 0, 0);
      __builtin_amdgcn_global_load_async_to_lds_b128(as1(g + 8),
                                                     as3(&sA[r][c + 8]), 0, 0);
    }
    { /* B tile 64x32: one async b128 transfer per thread */
      int idx = tid * 8;
      int r = idx >> 5, c = idx & 31;
      const bf16_t* g = W + (size_t)(n0 + r) * Kdim + kt + c;
      __builtin_amdgcn_global_load_async_to_lds_b128(as1(g),
                                                     as3(&sB[r][c]), 0, 0);
    }
    /* barrier does NOT imply counter waits: drain ASYNCcnt first */
    __builtin_amdgcn_s_wait_asynccnt(0);
#else
    { /* A tile 128x32: 16 bf16 per thread (2x b128) */
      int idx = tid * 16;
      int r = idx >> 5, c = idx & 31;
      const uint4* src = (const uint4*)(A + (size_t)(m0 + r) * Kdim + kt + c);
      uint4 d0 = src[0];
      uint4 d1 = src[1];
      *(uint4*)&sA[r][c]     = d0;
      *(uint4*)&sA[r][c + 8] = d1;
    }
    { /* W tile 64x32 (rows are N-indices, K contiguous): 8 bf16/thread */
      int idx = tid * 8;
      int r = idx >> 5, c = idx & 31;
      const uint4* src = (const uint4*)(W + (size_t)(n0 + r) * Kdim + kt + c);
      *(uint4*)&sB[r][c] = src[0];
    }
#endif
    __syncthreads();

    /* A fragment (16x32, ISA A-layout): two 8-elem K runs per lane */
    v16bf a;
    {
      const char* p = (const char*)&sA[wave * 16 + (lane & 15)][0];
      int off = 16 * (lane >> 4);
      FragU f;
      f.u[0] = *(const uint4*)(p + off);
      f.u[1] = *(const uint4*)(p + off + 32);
      a = f.v;
    }
#pragma unroll
    for (int g = 0; g < 4; ++g) {
      /* B fragment (32x16, ISA B-layout): two 8-elem K runs per lane */
      const char* p = (const char*)&sB[g * 16 + (lane & 15)][0];
      int off = 16 * (lane >> 4);
      FragU f;
      f.u[0] = *(const uint4*)(p + off);
      f.u[1] = *(const uint4*)(p + off + 32);
      acc[g] = __builtin_amdgcn_wmma_f32_16x16x32_bf16(
          false, a, false, f.v, (short)0, acc[g], false, false);
    }
  }

  /* epilogue; C-layout: VGPR r -> row r + 8*(lane>=16), col = lane&15 */
  const int col = lane & 15;
  const int rb  = (lane >> 4) * 8;
  if (mode == 0) {
#pragma unroll
    for (int g = 0; g < 4; ++g) {
      int n = n0 + g * 16 + col;
      float bv = bias[n];
#pragma unroll
      for (int r = 0; r < 8; ++r) {
        int m = m0 + wave * 16 + rb + r;
        outF[(size_t)m * CDIM + n] = acc[g][r] + bv;
      }
    }
  } else {
    const int which = n0 / CDIM;          /* 0=Q 1=K 2=V (tile never straddles) */
    const int h     = (n0 % CDIM) / HD;   /* tile never straddles a head */
#pragma unroll
    for (int g = 0; g < 4; ++g) {
      int d = g * 16 + col;               /* n0 is 64-aligned inside head */
#pragma unroll
      for (int r = 0; r < 8; ++r) {
        int m = m0 + wave * 16 + rb + r;
        int b = m >> 10, t = m & 1023;
        bf16_t v = (bf16_t)acc[g][r];
        size_t head = (size_t)(b * NH + h);
        if (which == 0)      qb[(head * NSEQ + t) * HD + d]   = v;
        else if (which == 1) kb[(head * NSEQ + t) * HD + d]   = v;
        else                 vtb[(head * HD + d) * NSEQ + t]  = v;  /* V^T */
      }
    }
  }
}

/* ----------------------------------------------------------
 * Flash attention: one (b,h) per blockIdx.y, 128 queries per
 * block (8 waves x 16 rows). Keys processed in tiles of 32.
 * S = Q K^T * scale -> online softmax -> O += P V (via V^T).
 * ---------------------------------------------------------- */
__global__ void __launch_bounds__(256)
flash_attn_kernel(const bf16_t* __restrict__ qb, const bf16_t* __restrict__ kb,
                  const bf16_t* __restrict__ vtb, bf16_t* __restrict__ attnb) {
  __shared__ bf16_t sP[8][16][40];   /* per-wave P transpose slab */

  const int tid  = threadIdx.x;
  const int lane = tid & 31;
  const int wave = tid >> 5;
  const int bh   = blockIdx.y;          /* 0..95 */
  const int b    = bh / NH, h = bh % NH;
  const int q0   = blockIdx.x * 128 + wave * 16;

  const bf16_t* Qp = qb  + (size_t)bh * NSEQ * HD;
  const bf16_t* Kp = kb  + (size_t)bh * NSEQ * HD;
  const bf16_t* Vt = vtb + (size_t)bh * HD * NSEQ;

  /* Q fragments once: contraction D=64 -> two 32-chunks */
  v16bf aq[2];
  {
    const char* p = (const char*)(Qp + (size_t)(q0 + (lane & 15)) * HD);
#pragma unroll
    for (int kc = 0; kc < 2; ++kc) {
      int off = kc * 64 + 16 * (lane >> 4);
      FragU f;
      f.u[0] = *(const uint4*)(p + off);
      f.u[1] = *(const uint4*)(p + off + 32);
      aq[kc] = f.v;
    }
  }

  float run_m[8], run_l[8];
  v8f oacc[4];
#pragma unroll
  for (int r = 0; r < 8; ++r) { run_m[r] = -3.0e38f; run_l[r] = 0.0f; }
#pragma unroll
  for (int g = 0; g < 4; ++g)
#pragma unroll
    for (int r = 0; r < 8; ++r) oacc[g][r] = 0.0f;

  const float scale = 0.125f;  /* 1/sqrt(64) */

  for (int key0 = 0; key0 < NSEQ; key0 += 32) {
    if (key0 + 32 < NSEQ)
      __builtin_prefetch(Kp + (size_t)(key0 + 32) * HD, 0, 1);

    /* ---- S = Q K^T over this 32-key tile: 2 col-groups x 2 K-chunks ---- */
    v8f s[2];
#pragma unroll
    for (int g = 0; g < 2; ++g)
#pragma unroll
      for (int r = 0; r < 8; ++r) s[g][r] = 0.0f;

#pragma unroll
    for (int g = 0; g < 2; ++g) {
      const char* p =
          (const char*)(Kp + (size_t)(key0 + g * 16 + (lane & 15)) * HD);
#pragma unroll
      for (int kc = 0; kc < 2; ++kc) {
        int off = kc * 64 + 16 * (lane >> 4);
        FragU f;
        f.u[0] = *(const uint4*)(p + off);
        f.u[1] = *(const uint4*)(p + off + 32);
        s[g] = __builtin_amdgcn_wmma_f32_16x16x32_bf16(
            false, aq[kc], false, f.v, (short)0, s[g], false, false);
      }
    }

    /* ---- online softmax (rows striped: VGPR r, cols across 16 lanes) ---- */
    float corr[8];
#pragma unroll
    for (int r = 0; r < 8; ++r) {
      float s0 = s[0][r] * scale;
      float s1 = s[1][r] * scale;
      float mloc = fmaxf(s0, s1);
#pragma unroll
      for (int off = 1; off < 16; off <<= 1)
        mloc = fmaxf(mloc, __shfl_xor(mloc, off, 32));
      float mnew = fmaxf(run_m[r], mloc);
      float c = __expf(run_m[r] - mnew);
      s0 = __expf(s0 - mnew);
      s1 = __expf(s1 - mnew);
      float lloc = s0 + s1;
#pragma unroll
      for (int off = 1; off < 16; off <<= 1)
        lloc += __shfl_xor(lloc, off, 32);
      run_l[r] = run_l[r] * c + lloc;
      run_m[r] = mnew;
      corr[r]  = c;
      s[0][r] = s0;
      s[1][r] = s1;
    }
#pragma unroll
    for (int g = 0; g < 4; ++g)
#pragma unroll
      for (int r = 0; r < 8; ++r) oacc[g][r] *= corr[r];

    /* ---- P: C-layout -> A-layout via per-wave LDS slab ---- */
    {
      const int col = lane & 15, rbb = (lane >> 4) * 8;
#pragma unroll
      for (int r = 0; r < 8; ++r) {
        sP[wave][rbb + r][col]      = (bf16_t)s[0][r];
        sP[wave][rbb + r][16 + col] = (bf16_t)s[1][r];
      }
    }
    v16bf ap;
    {
      const char* p = (const char*)&sP[wave][lane & 15][0];
      int off = 16 * (lane >> 4);
      FragU f;
      f.u[0] = *(const uint4*)(p + off);
      f.u[1] = *(const uint4*)(p + off + 32);
      ap = f.v;
    }

    /* ---- O += P * V  (V^T rows are d, K=keys contiguous) ---- */
#pragma unroll
    for (int g = 0; g < 4; ++g) {
      const char* p =
          (const char*)(Vt + (size_t)(g * 16 + (lane & 15)) * NSEQ + key0);
      int off = 16 * (lane >> 4);
      FragU f;
      f.u[0] = *(const uint4*)(p + off);
      f.u[1] = *(const uint4*)(p + off + 32);
      oacc[g] = __builtin_amdgcn_wmma_f32_16x16x32_bf16(
          false, ap, false, f.v, (short)0, oacc[g], false, false);
    }
  }

  /* ---- normalize + store to (B,N,C) bf16 for projection GEMM ---- */
  const int col = lane & 15, rb = (lane >> 4) * 8;
#pragma unroll
  for (int g = 0; g < 4; ++g) {
    int d = g * 16 + col;
#pragma unroll
    for (int r = 0; r < 8; ++r) {
      int t = q0 + rb + r;
      float inv = 1.0f / run_l[r];
      attnb[((size_t)(b * NSEQ + t)) * CDIM + h * HD + d] =
          (bf16_t)(oacc[g][r] * inv);
    }
  }
}

/* ---------------------------------------------------------- */
extern "C" void kernel_launch(void* const* d_in, const int* in_sizes, int n_in,
                              void* d_out, int out_size, void* d_ws,
                              size_t ws_size, hipStream_t stream) {
  (void)in_sizes; (void)n_in; (void)out_size; (void)ws_size;
  const float* x      = (const float*)d_in[0];
  const float* qkv_w  = (const float*)d_in[1];
  const float* proj_w = (const float*)d_in[2];
  const float* proj_b = (const float*)d_in[3];
  float* out = (float*)d_out;

  char* ws = (char*)d_ws;
  size_t off = 0;
  auto alloc = [&](size_t bytes) {
    void* p = ws + off;
    off = (off + bytes + 255) & ~(size_t)255;
    return p;
  };
  bf16_t* xb     = (bf16_t*)alloc((size_t)MTOK * CDIM * 2);
  bf16_t* wqkvb  = (bf16_t*)alloc((size_t)QKVN * CDIM * 2);
  bf16_t* wprojb = (bf16_t*)alloc((size_t)CDIM * CDIM * 2);
  bf16_t* qb     = (bf16_t*)alloc((size_t)BD * NH * NSEQ * HD * 2);
  bf16_t* kbuf   = (bf16_t*)alloc((size_t)BD * NH * NSEQ * HD * 2);
  bf16_t* vtb    = (bf16_t*)alloc((size_t)BD * NH * HD * NSEQ * 2);
  bf16_t* attnb  = (bf16_t*)alloc((size_t)MTOK * CDIM * 2);

  /* fp32 -> bf16 */
  f32_to_bf16_kernel<<<(MTOK * CDIM + 255) / 256, 256, 0, stream>>>(
      x, xb, MTOK * CDIM);
  f32_to_bf16_kernel<<<(QKVN * CDIM + 255) / 256, 256, 0, stream>>>(
      qkv_w, wqkvb, QKVN * CDIM);
  f32_to_bf16_kernel<<<(CDIM * CDIM + 255) / 256, 256, 0, stream>>>(
      proj_w, wprojb, CDIM * CDIM);

  /* QKV projection, scatter to per-head Q, K, V^T */
  dim3 g1(QKVN / 64, MTOK / 128);
  gemm_bf16_kernel<<<g1, 256, 0, stream>>>(xb, wqkvb, CDIM, 1, nullptr,
                                           nullptr, qb, kbuf, vtb);

  /* flash attention */
  dim3 g2(NSEQ / 128, BD * NH);
  flash_attn_kernel<<<g2, 256, 0, stream>>>(qb, kbuf, vtb, attnb);

  /* output projection + bias */
  dim3 g3(CDIM / 64, MTOK / 128);
  gemm_bf16_kernel<<<g3, 256, 0, stream>>>(attnb, wprojb, CDIM, 0, out,
                                           proj_b, nullptr, nullptr, nullptr);
}
